// LearnerKnowledge_Aggregator_12120397709895
// MI455X (gfx1250) — matrix operand order repaired and checked by reading, not compiled, not run on hardware
//
#include <hip/hip_runtime.h>
#include <hip/hip_bf16.h>

typedef _Float16 v16h __attribute__((ext_vector_type(16)));
typedef _Float16 v8h  __attribute__((ext_vector_type(8)));
typedef _Float16 v4h  __attribute__((ext_vector_type(4)));
typedef float    v8f  __attribute__((ext_vector_type(8)));
typedef float    v4f  __attribute__((ext_vector_type(4)));

union AF { v16h v; v8h h[2]; };

#define KN 50
#define DN 64

// One workgroup (128 threads = 4 wave32) per node b.
//   wave w owns output N-tile [w*16, w*16+16) of both GEMMs.
__global__ __launch_bounds__(128)
void gat_agg_kernel(const int* __restrict__ nodes,
                    const int* __restrict__ history,
                    const int* __restrict__ degrees,
                    const float* __restrict__ u_to_e,
                    const float* __restrict__ v_to_e,
                    const float* __restrict__ att1_w, const float* __restrict__ att1_b,
                    const float* __restrict__ att2_w, const float* __restrict__ att2_b,
                    const float* __restrict__ att3_w, const float* __restrict__ att3_b,
                    float* __restrict__ out)
{
    __shared__ _Float16 Xh [64 * 128]; // [neigh | self] rows, f16
    __shared__ _Float16 H1h[64 * 64];  // relu(X @ W1^T + b1), f16
    __shared__ float    H2f[64 * 64];  // relu(H1 @ W2^T + b2), f32
    __shared__ float    W3s[64];
    __shared__ float    Ls [64];
    __shared__ float    As [64];
    __shared__ int      HsI[64];

    const int b    = blockIdx.x;
    const int t    = threadIdx.x;
    const int w    = t >> 5;        // wave id 0..3  == N-tile
    const int lane = t & 31;
    const int nlo  = lane & 15;
    const int hi   = (lane & 16) ? 1 : 0;

    const int node = nodes[b];
    int deg = degrees[b];
    deg = deg < KN ? deg : KN;
    deg = deg < 1 ? 1 : deg;

    if (t < KN) HsI[t] = history[(size_t)b * KN + t];
    if (t >= KN && t < 64) HsI[t] = 0;
    if (t < 64) W3s[t] = att3_w[t];
    __syncthreads();

    // ---- Stage X = [v_to_e[hist[m]] | u_to_e[node]] as f16; rows 50..63 zero.
    {
        const int c = lane * 4;                  // 32 lanes * 4 floats = 128 cols
        for (int m = w; m < 64; m += 4) {        // 16 rows per wave
            v4f val = {0.f, 0.f, 0.f, 0.f};
            if (m < KN) {
                if (c < DN) val = *(const v4f*)&v_to_e[(size_t)HsI[m] * DN + c];
                else        val = *(const v4f*)&u_to_e[(size_t)node   * DN + (c - DN)];
            }
            v4h h;
            h[0] = (_Float16)val[0]; h[1] = (_Float16)val[1];
            h[2] = (_Float16)val[2]; h[3] = (_Float16)val[3];
            *(v4h*)&Xh[m * 128 + c] = h;
        }
    }

    // ---- Per-wave weight B-fragments (B[k][n] = W[n][k], n = d fixed per lane).
    // 16-bit B 32x16 layout: lane half selects K runs {0-7,16-23} / {8-15,24-31}.
    const int d = w * 16 + nlo;                  // output column this lane owns
    AF b1f[4];
    #pragma unroll
    for (int kt = 0; kt < 4; ++kt) {
        const float* src = &att1_w[(size_t)d * 128 + kt * 32 + hi * 8];
        #pragma unroll
        for (int j = 0; j < 16; ++j) b1f[kt].v[j] = (_Float16)src[j + (j >= 8 ? 8 : 0)];
    }
    AF b2f[2];
    #pragma unroll
    for (int kt = 0; kt < 2; ++kt) {
        const float* src = &att2_w[(size_t)d * 64 + kt * 32 + hi * 8];
        #pragma unroll
        for (int j = 0; j < 16; ++j) b2f[kt].v[j] = (_Float16)src[j + (j >= 8 ? 8 : 0)];
    }
    const float b1n = att1_b[d];
    const float b2n = att2_b[d];

    __syncthreads();

    // ---- GEMM1: H1 = relu(X @ W1^T + b1).  M=64, K=128, this wave: N-tile w.
    // 16-bit A 16x32 layout: lane half selects K runs {0-7,16-23} / {8-15,24-31}.
    #pragma unroll
    for (int mt = 0; mt < 4; ++mt) {
        v8f acc = {};
        const int m = mt * 16 + nlo;
        #pragma unroll
        for (int kt = 0; kt < 4; ++kt) {
            AF a;
            const int f0 = kt * 32 + hi * 8;
            a.h[0] = *(const v8h*)&Xh[m * 128 + f0];
            a.h[1] = *(const v8h*)&Xh[m * 128 + f0 + 16];
            acc = __builtin_amdgcn_wmma_f32_16x16x32_f16(
                false, a.v, false, b1f[kt].v, (short)0, acc, false, false);
        }
        const int mb = mt * 16 + hi * 8;         // C/D: VGPR r -> M = r (+8 hi half)
        #pragma unroll
        for (int r = 0; r < 8; ++r) {
            float vv = acc[r] + b1n; vv = vv > 0.f ? vv : 0.f;
            H1h[(mb + r) * 64 + d] = (_Float16)vv;
        }
    }
    __syncthreads();

    // ---- GEMM2: H2 = relu(H1 @ W2^T + b2).  M=64, K=64.
    #pragma unroll
    for (int mt = 0; mt < 4; ++mt) {
        v8f acc = {};
        const int m = mt * 16 + nlo;
        #pragma unroll
        for (int kt = 0; kt < 2; ++kt) {
            AF a;
            const int f0 = kt * 32 + hi * 8;
            a.h[0] = *(const v8h*)&H1h[m * 64 + f0];
            a.h[1] = *(const v8h*)&H1h[m * 64 + f0 + 16];
            acc = __builtin_amdgcn_wmma_f32_16x16x32_f16(
                false, a.v, false, b2f[kt].v, (short)0, acc, false, false);
        }
        const int mb = mt * 16 + hi * 8;
        #pragma unroll
        for (int r = 0; r < 8; ++r) {
            float vv = acc[r] + b2n; vv = vv > 0.f ? vv : 0.f;
            H2f[(mb + r) * 64 + d] = vv;
        }
    }
    __syncthreads();

    // ---- logits[m] = H2[m,:] . w3 + b3
    if (t < 64) {
        float s = att3_b[0];
        #pragma unroll 8
        for (int n = 0; n < 64; ++n) s += H2f[t * 64 + n] * W3s[n];
        Ls[t] = s;
    }
    __syncthreads();

    // ---- ragged softmax over the first `deg` neighbors (redundant per lane, tiny)
    if (t < 64) {
        float mx = -1e30f;
        for (int m = 0; m < deg; ++m) mx = fmaxf(mx, Ls[m]);
        float s = 0.f;
        for (int m = 0; m < deg; ++m) s += __expf(Ls[m] - mx);
        As[t] = (t < deg) ? __expf(Ls[t] - mx) / s : 0.f;
    }
    __syncthreads();

    // ---- out[b,d] = sum_m att[m] * v_to_e[hist[m], d]   (f32, L2-resident table)
    if (t < 64) {
        float acc = 0.f;
        for (int m = 0; m < deg; ++m)
            acc += As[m] * v_to_e[(size_t)HsI[m] * DN + t];
        out[(size_t)b * DN + t] = acc;
    }
}

extern "C" void kernel_launch(void* const* d_in, const int* in_sizes, int n_in,
                              void* d_out, int out_size, void* d_ws, size_t ws_size,
                              hipStream_t stream) {
    (void)n_in; (void)d_ws; (void)ws_size; (void)out_size;
    const int*   nodes   = (const int*)  d_in[0];
    const int*   history = (const int*)  d_in[1];
    const int*   degrees = (const int*)  d_in[2];
    const float* u_to_e  = (const float*)d_in[3];
    const float* v_to_e  = (const float*)d_in[4];
    const float* att1_w  = (const float*)d_in[5];
    const float* att1_b  = (const float*)d_in[6];
    const float* att2_w  = (const float*)d_in[7];
    const float* att2_b  = (const float*)d_in[8];
    const float* att3_w  = (const float*)d_in[9];
    const float* att3_b  = (const float*)d_in[10];
    float* out = (float*)d_out;

    const int B = in_sizes[0];   // 8192 nodes, one block each
    gat_agg_kernel<<<B, 128, 0, stream>>>(nodes, history, degrees, u_to_e, v_to_e,
                                          att1_w, att1_b, att2_w, att2_b,
                                          att3_w, att3_b, out);
}
